// MultiHeadAttentionBlock_54649163875068
// MI455X (gfx1250) — matrix operand order repaired and checked by reading, not compiled, not run on hardware
//
#include <hip/hip_runtime.h>
#include <hip/hip_bf16.h>

#define DM   768
#define NH   12
#define DK   64
#define SQ   2048
#define MTOT 4096            // B*S
#define WSZ  (768*768)
#define LROW 72              // padded LDS row stride (elements) for 64-elem rows

typedef __bf16 bf16_t;
typedef __attribute__((ext_vector_type(16))) __bf16 v16bf;
typedef __attribute__((ext_vector_type(8)))  __bf16 v8bf;
typedef __attribute__((ext_vector_type(8)))  float  v8f;
typedef __attribute__((ext_vector_type(4)))  float  v4f;
typedef __attribute__((ext_vector_type(4)))  unsigned int u32x4;
typedef __attribute__((ext_vector_type(8)))  int i32x8;
typedef __attribute__((ext_vector_type(4)))  int i32x4;

union V16U { v16bf v; v8bf h[2]; };

// A/B fragment: two contiguous 8-element (16B) bf16 chunks.
__device__ __forceinline__ v16bf ld_frag(const bf16_t* c0, const bf16_t* c1) {
  V16U u;
  u.h[0] = *(const v8bf*)c0;
  u.h[1] = *(const v8bf*)c1;
  return u.v;
}

// A fragment built from fp32 source (8+8 contiguous floats), converted to bf16.
__device__ __forceinline__ v16bf cvt_frag_f32(const float* c0, const float* c1) {
  v4f a0 = *(const v4f*)c0, a1 = *(const v4f*)(c0 + 4);
  v4f b0 = *(const v4f*)c1, b1 = *(const v4f*)(c1 + 4);
  V16U u;
#pragma unroll
  for (int i = 0; i < 4; ++i) {
    u.v[i]      = (bf16_t)a0[i];
    u.v[4 + i]  = (bf16_t)a1[i];
    u.v[8 + i]  = (bf16_t)b0[i];
    u.v[12 + i] = (bf16_t)b1[i];
  }
  return u.v;
}

__device__ __forceinline__ v8f wmma_bf16(v16bf a, v16bf b, v8f c) {
  return __builtin_amdgcn_wmma_f32_16x16x32_bf16(false, a, false, b, (short)0, c,
                                                 false, false);
}

// ---------------------------------------------------------------------------
// TDM: DMA a 64x64 bf16 tile (global row stride 768 elements) into LDS.
// Pad feature (§8.4): pad_enable=1, pad_interval=4 (every 32 DWORDs = one
// 128B row), pad_amount=3 (4 DWORDs = 16B) -> LDS row stride 144B (72 elems),
// which makes the 16-lane ds_load_b128 fragment reads bank-conflict-free.
//   group0: count=1, lds_addr, global_addr[56:0], type=2
//   group1: data_size=1(2B), pad cfg, tensor_dim0/1=768, tile_dim0/1=64,
//           tensor_dim0_stride=768
// Tracked by TENSORcnt (s_wait_tensorcnt).  6-arg builtin (clang-23 lane).
// ---------------------------------------------------------------------------
__device__ __forceinline__ void tdm_load_tile_64x64(const bf16_t* gsrc,
                                                    unsigned lds_off) {
  unsigned long long ga = (unsigned long long)(uintptr_t)gsrc;
  u32x4 g0;
  g0[0] = 1u;                                   // count=1, user desc
  g0[1] = lds_off;                              // lds_addr (bytes)
  g0[2] = (unsigned)(ga & 0xffffffffu);         // global_addr[31:0]
  g0[3] = (unsigned)((ga >> 32) & 0x01ffffffu)  // global_addr[56:32]
          | 0x80000000u;                        // type=2 ("image")
  i32x8 g1;
  g1[0] = (1 << 16)              // workgroup_mask=0, data_size=1 (2 bytes)
        | (1 << 20)              // pad_enable
        | (4 << 22)              // pad_interval: every 32 DWORDs (128B)
        | (3 << 25);             // pad_amount: 4 DWORDs (16B)
  g1[1] = (int)(768u << 16);     // tensor_dim0[15:0] in bits 63:48
  g1[2] = (int)(768u << 16);     // tensor_dim0[31:16]=0 | tensor_dim1[15:0]
  g1[3] = 64 << 16;              // tensor_dim1[31:16]=0 | tile_dim0=64
  g1[4] = 64;                    // tile_dim1=64, tile_dim2=0
  g1[5] = 768;                   // tensor_dim0_stride[31:0]
  g1[6] = 0;                     // stride hi / dim1_stride lo
  g1[7] = 0;                     // dim1_stride hi (unused for 2D)
  i32x4 gz4 = {0, 0, 0, 0};      // groups 2/3 unused (2D tensor)
  i32x8 gz8 = {0, 0, 0, 0, 0, 0, 0, 0};
  __builtin_amdgcn_tensor_load_to_lds(g0, g1, gz4, gz4, gz8, 0);
}

// ---------------------------------------------------------------------------
// Kernel 1: convert the four 768x768 fp32 weight matrices to bf16 (row-major,
// i.e. already in WMMA B-fragment-friendly layout for x @ W^T).
// ---------------------------------------------------------------------------
__global__ void mha_convert_weights(const float* __restrict__ wq,
                                    const float* __restrict__ wk,
                                    const float* __restrict__ wv,
                                    const float* __restrict__ wo,
                                    bf16_t* __restrict__ wbf) {
  int z = blockIdx.y;
  const float* src = (z == 0) ? wq : (z == 1) ? wk : (z == 2) ? wv : wo;
  size_t i = (size_t)blockIdx.x * blockDim.x + threadIdx.x;
  wbf[(size_t)z * WSZ + i] = (bf16_t)src[i];
}

// ---------------------------------------------------------------------------
// Kernel 2: fused Q/K/V projections.  Y = X @ W^T + b, X:[4096,768] fp32.
// z=0 -> Q (bf16 row-major, pre-scaled by log2(e)/sqrt(Dk))
// z=1 -> K (bf16 row-major)
// z=2 -> V (bf16 TRANSPOSED:  VhT[(b*768 + col)][s],  col = h*64+d)
// One wave computes a 16x64 strip; 8 waves stack in M -> WG covers 128x64.
// 64x64 weight tiles staged via TDM (padded rows), double-buffered: 8 WMMA
// between barrier pairs, next DMA overlapped with compute.
// ---------------------------------------------------------------------------
__global__ void mha_proj_qkv(const float* __restrict__ q,
                             const float* __restrict__ k,
                             const float* __restrict__ v,
                             const bf16_t* __restrict__ wbf,
                             const float* __restrict__ bq,
                             const float* __restrict__ bk,
                             const float* __restrict__ bv,
                             bf16_t* __restrict__ Yq,
                             bf16_t* __restrict__ Yk,
                             bf16_t* __restrict__ VhT) {
  __shared__ __attribute__((aligned(16))) bf16_t wtile[2][64 * LROW];

  const int lane  = threadIdx.x & 31;
  const int w     = threadIdx.x >> 5;
  const int lhalf = lane >> 4;
  const int l16   = lane & 15;
  const int mbase = blockIdx.x * 128 + w * 16;
  const int nbase = blockIdx.y * 64;
  const int z     = blockIdx.z;

  const float*  X    = (z == 0) ? q : (z == 1) ? k : v;
  const float*  bias = (z == 0) ? bq : (z == 1) ? bk : bv;
  const bf16_t* W    = wbf + (size_t)z * WSZ;
  const bf16_t* Wrow = W + (size_t)nbase * DM;

  const unsigned lds0 = (unsigned)(uintptr_t)&wtile[0][0];
  const unsigned lds1 = (unsigned)(uintptr_t)&wtile[1][0];

  const float* arow = X + (size_t)(mbase + l16) * DM;

  v8f zero = {0.f, 0.f, 0.f, 0.f, 0.f, 0.f, 0.f, 0.f};
  v8f acc[4] = {zero, zero, zero, zero};

  if (w == 0) tdm_load_tile_64x64(Wrow, lds0);   // prime buffer 0

  for (int ki = 0; ki < DM / 64; ++ki) {
    const int kk = ki * 64;
    const bf16_t* buf = &wtile[ki & 1][0];
    if (w == 0) __builtin_amdgcn_s_wait_tensorcnt(0);
    __syncthreads();                             // tile kk is visible
    if (w == 0 && ki + 1 < DM / 64)              // overlap next DMA w/ compute
      tdm_load_tile_64x64(Wrow + (kk + 64), ((ki + 1) & 1) ? lds1 : lds0);

    v16bf a0 = cvt_frag_f32(arow + kk + lhalf * 16, arow + kk + 8 + lhalf * 16);
    v16bf a1 =
        cvt_frag_f32(arow + kk + 32 + lhalf * 16, arow + kk + 40 + lhalf * 16);
#pragma unroll
    for (int t = 0; t < 4; ++t) {
      const bf16_t* wp = buf + (t * 16 + l16) * LROW;
      acc[t] = wmma_bf16(a0, ld_frag(wp + lhalf * 16, wp + lhalf * 16 + 8),
                         acc[t]);
      acc[t] = wmma_bf16(
          a1, ld_frag(wp + 32 + lhalf * 16, wp + 40 + lhalf * 16), acc[t]);
    }
    __syncthreads();                             // done reading this buffer
  }

  const int b    = mbase >> 11;       // / 2048
  const int srow = mbase & (SQ - 1);  // % 2048
  const float qscale = 0.18033688011112042f;  // log2(e)/sqrt(64)

#pragma unroll
  for (int t = 0; t < 4; ++t) {
    const int   n  = nbase + t * 16 + l16;
    const float bv_ = bias[n];
    if (z == 2) {
      v8bf pack;
#pragma unroll
      for (int r = 0; r < 8; ++r) pack[r] = (bf16_t)(acc[t][r] + bv_);
      *(v8bf*)(VhT + ((size_t)(b * DM + n)) * SQ + srow + lhalf * 8) = pack;
    } else if (z == 0) {
#pragma unroll
      for (int r = 0; r < 8; ++r)
        Yq[(size_t)(mbase + 8 * lhalf + r) * DM + n] =
            (bf16_t)((acc[t][r] + bv_) * qscale);
    } else {
#pragma unroll
      for (int r = 0; r < 8; ++r)
        Yk[(size_t)(mbase + 8 * lhalf + r) * DM + n] =
            (bf16_t)(acc[t][r] + bv_);
    }
  }
}

// ---------------------------------------------------------------------------
// Kernel 3: causal flash attention.  One wave = 16 queries x Dk=64 for one
// (b,h).  Keys processed 32 at a time: 4 WMMA for scores, online softmax in
// C-layout (16-lane shfl_xor reductions), P re-laid C->A via padded LDS,
// 4 WMMA for P @ V.  Output written bf16 into xcat[4096][768].
// ---------------------------------------------------------------------------
__global__ void mha_flash_attn(const bf16_t* __restrict__ Qb,
                               const bf16_t* __restrict__ Kb,
                               const bf16_t* __restrict__ VhT,
                               bf16_t* __restrict__ xcat) {
  __shared__ __attribute__((aligned(16))) bf16_t lds_p[8 * 16 * 40];

  const int lane  = threadIdx.x & 31;
  const int w     = threadIdx.x >> 5;
  const int lhalf = lane >> 4;
  const int l16   = lane & 15;
  const int qtile = blockIdx.x * 128 + w * 16;
  const int b     = blockIdx.y / NH;
  const int h     = blockIdx.y % NH;

  bf16_t* ldsw = &lds_p[w * 640];

  // Q fragments (16 queries x Dk=64 -> two k32 A fragments), already scaled.
  const bf16_t* qrow = Qb + (size_t)(b * SQ + qtile + l16) * DM + h * DK;
  v16bf aq0 = ld_frag(qrow + lhalf * 16,      qrow + 8 + lhalf * 16);
  v16bf aq1 = ld_frag(qrow + 32 + lhalf * 16, qrow + 40 + lhalf * 16);

  const bf16_t* Kbase = Kb + (size_t)(b * SQ) * DM + h * DK;
  const bf16_t* Vbase = VhT + (size_t)(b * DM + h * DK) * SQ;

  v8f zero = {0.f, 0.f, 0.f, 0.f, 0.f, 0.f, 0.f, 0.f};
  v8f o[4] = {zero, zero, zero, zero};
  float m8[8], l8[8];
#pragma unroll
  for (int r = 0; r < 8; ++r) { m8[r] = -1e30f; l8[r] = 0.f; }

  const int nblocks = (qtile + 15) / 32 + 1;
  for (int j = 0; j < nblocks; ++j) {
    const int kbase = j * 32;
    const bf16_t* k0p = Kbase + (size_t)(kbase + l16) * DM;
    const bf16_t* k1p = Kbase + (size_t)(kbase + 16 + l16) * DM;

    if (j + 1 < nblocks) {  // prefetch next K block (global_prefetch_b8)
      __builtin_prefetch(Kbase + (size_t)(kbase + 32 + l16) * DM, 0, 1);
      __builtin_prefetch(Kbase + (size_t)(kbase + 48 + l16) * DM, 0, 1);
    }

    // scores: 16 queries x 32 keys, K-dim = Dk = 64 (two k32 steps each)
    v8f s0 = zero, s1 = zero;
    s0 = wmma_bf16(aq0, ld_frag(k0p + lhalf * 16, k0p + lhalf * 16 + 8), s0);
    s0 = wmma_bf16(aq1, ld_frag(k0p + 32 + lhalf * 16, k0p + 40 + lhalf * 16), s0);
    s1 = wmma_bf16(aq0, ld_frag(k1p + lhalf * 16, k1p + lhalf * 16 + 8), s1);
    s1 = wmma_bf16(aq1, ld_frag(k1p + 32 + lhalf * 16, k1p + 40 + lhalf * 16), s1);

    if (kbase + 31 > qtile) {  // causal mask (diagonal blocks only)
#pragma unroll
      for (int r = 0; r < 8; ++r) {
        const int qrow_i = qtile + 8 * lhalf + r;
        if (kbase + l16 > qrow_i)      s0[r] = -1e30f;
        if (kbase + 16 + l16 > qrow_i) s1[r] = -1e30f;
      }
    }

    // online softmax, row stats per C-layout row group (log2 domain)
    float es[8];
#pragma unroll
    for (int r = 0; r < 8; ++r) {
      float t0 = s0[r], t1 = s1[r];
      float mx = fmaxf(t0, t1);
#pragma unroll
      for (int off = 1; off < 16; off <<= 1)
        mx = fmaxf(mx, __shfl_xor(mx, off, 32));
      const float mn = fmaxf(m8[r], mx);
      const float e  = exp2f(m8[r] - mn);
      const float p0 = exp2f(t0 - mn);
      const float p1 = exp2f(t1 - mn);
      float rs = p0 + p1;
#pragma unroll
      for (int off = 1; off < 16; off <<= 1) rs += __shfl_xor(rs, off, 32);
      l8[r] = l8[r] * e + rs;
      m8[r] = mn;
      es[r] = e;
      const int row = 8 * lhalf + r;
      ldsw[row * 40 + l16]      = (bf16_t)p0;   // C-layout -> LDS row-major
      ldsw[row * 40 + 16 + l16] = (bf16_t)p1;
    }
#pragma unroll
    for (int t = 0; t < 4; ++t)
#pragma unroll
      for (int r = 0; r < 8; ++r) o[t][r] *= es[r];

    // wave-private LDS: wait for our own ds stores, then read back in A layout
    asm volatile("s_wait_dscnt 0" ::: "memory");
    const bf16_t* pr = &ldsw[l16 * 40];
    v16bf ap = ld_frag(pr + lhalf * 16, pr + 8 + lhalf * 16);

    // O += P @ V  (V stored transposed: B fragments contiguous)
#pragma unroll
    for (int t = 0; t < 4; ++t) {
      const bf16_t* vp =
          Vbase + (size_t)(t * 16 + l16) * SQ + kbase + lhalf * 16;
      o[t] = wmma_bf16(ap, ld_frag(vp, vp + 8), o[t]);
    }
  }

  // normalize and emit bf16 into xcat[(b*S+s)][h*64+d]
  float inv[8];
#pragma unroll
  for (int r = 0; r < 8; ++r) inv[r] = 1.0f / l8[r];
#pragma unroll
  for (int t = 0; t < 4; ++t) {
#pragma unroll
    for (int r = 0; r < 8; ++r) {
      xcat[(size_t)(b * SQ + qtile + 8 * lhalf + r) * DM + h * DK + t * 16 +
           l16] = (bf16_t)(o[t][r] * inv[r]);
    }
  }
}

// ---------------------------------------------------------------------------
// Kernel 4: output projection.  out = xcat @ wo^T + bo, fp32 result.
// 64x64 weight tiles staged via TDM (padded rows), double-buffered.
// ---------------------------------------------------------------------------
__global__ void mha_proj_out(const bf16_t* __restrict__ xcat,
                             const bf16_t* __restrict__ wob,
                             const float* __restrict__ bo,
                             float* __restrict__ out) {
  __shared__ __attribute__((aligned(16))) bf16_t wtile[2][64 * LROW];

  const int lane  = threadIdx.x & 31;
  const int w     = threadIdx.x >> 5;
  const int lhalf = lane >> 4;
  const int l16   = lane & 15;
  const int mbase = blockIdx.x * 128 + w * 16;
  const int nbase = blockIdx.y * 64;

  const bf16_t* Wrow = wob + (size_t)nbase * DM;
  const unsigned lds0 = (unsigned)(uintptr_t)&wtile[0][0];
  const unsigned lds1 = (unsigned)(uintptr_t)&wtile[1][0];

  const bf16_t* arow = xcat + (size_t)(mbase + l16) * DM;

  v8f zero = {0.f, 0.f, 0.f, 0.f, 0.f, 0.f, 0.f, 0.f};
  v8f acc[4] = {zero, zero, zero, zero};

  if (w == 0) tdm_load_tile_64x64(Wrow, lds0);

  for (int ki = 0; ki < DM / 64; ++ki) {
    const int kk = ki * 64;
    const bf16_t* buf = &wtile[ki & 1][0];
    if (w == 0) __builtin_amdgcn_s_wait_tensorcnt(0);
    __syncthreads();
    if (w == 0 && ki + 1 < DM / 64)
      tdm_load_tile_64x64(Wrow + (kk + 64), ((ki + 1) & 1) ? lds1 : lds0);

    v16bf a0 = ld_frag(arow + kk + lhalf * 16, arow + kk + 8 + lhalf * 16);
    v16bf a1 =
        ld_frag(arow + kk + 32 + lhalf * 16, arow + kk + 40 + lhalf * 16);
#pragma unroll
    for (int t = 0; t < 4; ++t) {
      const bf16_t* wp = buf + (t * 16 + l16) * LROW;
      acc[t] = wmma_bf16(a0, ld_frag(wp + lhalf * 16, wp + lhalf * 16 + 8),
                         acc[t]);
      acc[t] = wmma_bf16(
          a1, ld_frag(wp + 32 + lhalf * 16, wp + 40 + lhalf * 16), acc[t]);
    }
    __syncthreads();
  }

#pragma unroll
  for (int t = 0; t < 4; ++t) {
    const int   n   = nbase + t * 16 + l16;
    const float bv_ = bo[n];
#pragma unroll
    for (int r = 0; r < 8; ++r)
      out[(size_t)(mbase + 8 * lhalf + r) * DM + n] = acc[t][r] + bv_;
  }
}

// ---------------------------------------------------------------------------
extern "C" void kernel_launch(void* const* d_in, const int* in_sizes, int n_in,
                              void* d_out, int out_size, void* d_ws,
                              size_t ws_size, hipStream_t stream) {
  const float* q  = (const float*)d_in[0];
  const float* k  = (const float*)d_in[1];
  const float* v  = (const float*)d_in[2];
  // d_in[3] = causal mask, handled analytically
  const float* wq = (const float*)d_in[4];
  const float* bq = (const float*)d_in[5];
  const float* wk = (const float*)d_in[6];
  const float* bk = (const float*)d_in[7];
  const float* wv = (const float*)d_in[8];
  const float* bv = (const float*)d_in[9];
  const float* wo = (const float*)d_in[10];
  const float* bo = (const float*)d_in[11];

  bf16_t* ws   = (bf16_t*)d_ws;
  bf16_t* wbf  = ws;                       // 4 * 768*768 bf16 weights
  bf16_t* Yq   = ws + (size_t)4 * WSZ;     // [4096][768]
  bf16_t* Yk   = Yq + (size_t)MTOT * DM;   // [4096][768]
  bf16_t* VhT  = Yk + (size_t)MTOT * DM;   // [2][768][2048] transposed V
  bf16_t* xcat = VhT + (size_t)MTOT * DM;  // [4096][768] attention output
  float*  out  = (float*)d_out;

  mha_convert_weights<<<dim3(WSZ / 256, 4), 256, 0, stream>>>(wq, wk, wv, wo,
                                                              wbf);
  mha_proj_qkv<<<dim3(MTOT / 128, DM / 64, 3), 256, 0, stream>>>(
      q, k, v, wbf, bq, bk, bv, Yq, Yk, VhT);
  mha_flash_attn<<<dim3(SQ / 128, 2 * NH), 256, 0, stream>>>(Yq, Yk, VhT,
                                                             xcat);
  mha_proj_out<<<dim3(MTOT / 128, DM / 64), 256, 0, stream>>>(
      xcat, wbf + (size_t)3 * WSZ, bo, out);
}
